// K_Sparse_Autoencoder_90941637526128
// MI455X (gfx1250) — compile-verified, matrix operand order checked by reading, not run on hardware
//
#include <hip/hip_runtime.h>
#include <stdint.h>

typedef __attribute__((ext_vector_type(16))) __bf16 v16bf;
typedef __attribute__((ext_vector_type(8)))  float  v8f;

constexpr int BSZ  = 4096;
constexpr int DIN  = 4096;
constexpr int DBN  = 32768;
constexpr int KTOP = 64;

// ---------------------------------------------------------------------------
// fp32 -> bf16 (RNE) conversion, 4-wide
// ---------------------------------------------------------------------------
__global__ void cvt_bf16x4(const float4* __restrict__ in,
                           ushort4* __restrict__ out, long n4) {
  long i = blockIdx.x * (long)blockDim.x + threadIdx.x;
  long stride = (long)gridDim.x * blockDim.x;
  for (; i < n4; i += stride) {
    float4 f = in[i];
    ushort4 o;
    uint32_t u;
    u = __float_as_uint(f.x); o.x = (uint16_t)((u + 0x7FFFu + ((u >> 16) & 1u)) >> 16);
    u = __float_as_uint(f.y); o.y = (uint16_t)((u + 0x7FFFu + ((u >> 16) & 1u)) >> 16);
    u = __float_as_uint(f.z); o.z = (uint16_t)((u + 0x7FFFu + ((u >> 16) & 1u)) >> 16);
    u = __float_as_uint(f.w); o.w = (uint16_t)((u + 0x7FFFu + ((u >> 16) & 1u)) >> 16);
    out[i] = o;
  }
}

// ---------------------------------------------------------------------------
// Encode GEMM: a1[m,n] = sum_k x[m,k]*W[n,k] + b_enc[n]
// Block tile 128x128, 8 waves; wave tile 32(M) x 64(N) = 8 accumulators
// (2 A-frags x 4 B-frags -> 8 WMMAs per K-step of 32).
// Double-buffered LDS; global->LDS staging uses the CDNA5 async path
// (GLOBAL_LOAD_ASYNC_TO_LDS_B128, ASYNCcnt) so no VGPR staging and one
// s_wait_asynccnt + barrier per K-step.
// ---------------------------------------------------------------------------
#define BM 128
#define BN 128
#define BK 32

__device__ __forceinline__ void async_tile4(uint32_t lA0, uint32_t lA1,
                                            uint32_t lB0, uint32_t lB1,
                                            const uint16_t* gA0, const uint16_t* gA1,
                                            const uint16_t* gB0, const uint16_t* gB1) {
  asm volatile(
      "global_load_async_to_lds_b128 %0, %4, off\n\t"
      "global_load_async_to_lds_b128 %1, %5, off\n\t"
      "global_load_async_to_lds_b128 %2, %6, off\n\t"
      "global_load_async_to_lds_b128 %3, %7, off"
      :
      : "v"(lA0), "v"(lA1), "v"(lB0), "v"(lB1),
        "v"(gA0), "v"(gA1), "v"(gB0), "v"(gB1)
      : "memory");
}

__device__ __forceinline__ void wait_async0() {
  asm volatile("s_wait_asynccnt 0x0" ::: "memory");
}

__global__ __launch_bounds__(256) void encode_gemm(
    const uint16_t* __restrict__ X,    // [BSZ, DIN] bf16
    const uint16_t* __restrict__ W,    // [DBN, DIN] bf16
    const float*    __restrict__ benc, // [DBN]
    float*          __restrict__ A1)   // [BSZ, DBN]
{
  __shared__ __align__(16) uint16_t As[2][BM * BK];  // 2 x 8 KB
  __shared__ __align__(16) uint16_t Bs[2][BN * BK];  // 2 x 8 KB

  const int tid  = threadIdx.x;
  const int lane = tid & 31;
  const int wid  = tid >> 5;      // 0..7
  const int wm   = wid & 3;       // M sub-tile: 32*wm
  const int wn   = wid >> 2;      // N sub-tile: 64*wn
  const int mBase = blockIdx.y * BM;
  const int nBase = blockIdx.x * BN;

  const int khalf = lane >> 4;    // 0/1: which K half-run this lane owns
  const int ml    = lane & 15;    // m (or n) within a 16-tile

  // cooperative tile-load mapping: 128 rows x 64B; thread covers 2 rows x 16B
  const int seg = (tid & 3) * 8;          // bf16 elem offset in row
  const int r0  = tid >> 2;               // 0..63
  const uint16_t* xr0 = X + (size_t)(mBase + r0)      * DIN + seg;
  const uint16_t* xr1 = X + (size_t)(mBase + r0 + 64) * DIN + seg;
  const uint16_t* wr0 = W + (size_t)(nBase + r0)      * DIN + seg;
  const uint16_t* wr1 = W + (size_t)(nBase + r0 + 64) * DIN + seg;
  const int ldsOff0 = r0 * BK + seg;
  const int ldsOff1 = (r0 + 64) * BK + seg;

  // LDS byte addresses (flat-pointer low 32 bits == wave-relative LDS addr)
  const uint32_t lAs0_0 = (uint32_t)(uintptr_t)&As[0][ldsOff0];
  const uint32_t lAs0_1 = (uint32_t)(uintptr_t)&As[0][ldsOff1];
  const uint32_t lBs0_0 = (uint32_t)(uintptr_t)&Bs[0][ldsOff0];
  const uint32_t lBs0_1 = (uint32_t)(uintptr_t)&Bs[0][ldsOff1];
  const uint32_t lAs1_0 = (uint32_t)(uintptr_t)&As[1][ldsOff0];
  const uint32_t lAs1_1 = (uint32_t)(uintptr_t)&As[1][ldsOff1];
  const uint32_t lBs1_0 = (uint32_t)(uintptr_t)&Bs[1][ldsOff0];
  const uint32_t lBs1_1 = (uint32_t)(uintptr_t)&Bs[1][ldsOff1];

  // prologue: async-stage K-tile 0 into buffer 0
  async_tile4(lAs0_0, lAs0_1, lBs0_0, lBs0_1, xr0, xr1, wr0, wr1);
  wait_async0();
  __syncthreads();

  v8f c[2][4] = {};

  const int aFragOff0 = (wm * 32 + ml) * BK + khalf * 8;       // A frag rows +0..15
  const int aFragOff1 = (wm * 32 + 16 + ml) * BK + khalf * 8;  // A frag rows +16..31

  for (int k0 = 0; k0 < DIN; k0 += BK) {
    const int cur  = (k0 >> 5) & 1;
    const bool more = (k0 + BK) < DIN;

    // kick off next tile's async global->LDS copies into the other buffer
    if (more) {
      if (cur == 0)
        async_tile4(lAs1_0, lAs1_1, lBs1_0, lBs1_1,
                    xr0 + k0 + BK, xr1 + k0 + BK, wr0 + k0 + BK, wr1 + k0 + BK);
      else
        async_tile4(lAs0_0, lAs0_1, lBs0_0, lBs0_1,
                    xr0 + k0 + BK, xr1 + k0 + BK, wr0 + k0 + BK, wr1 + k0 + BK);
    }

    // A fragments (2) for this wave
    v16bf a0, a1;
    {
      const uint16_t* p0 = &As[cur][aFragOff0];
      const uint16_t* p1 = &As[cur][aFragOff1];
      uint4* ap0 = (uint4*)&a0;
      uint4* ap1 = (uint4*)&a1;
      ap0[0] = *(const uint4*)(p0);
      ap0[1] = *(const uint4*)(p0 + 16);
      ap1[0] = *(const uint4*)(p1);
      ap1[1] = *(const uint4*)(p1 + 16);
    }
#pragma unroll
    for (int t = 0; t < 4; ++t) {
      v16bf b;  // same per-lane layout as A (W row-major over k: no transpose)
      const uint16_t* q = &Bs[cur][(wn * 64 + t * 16 + ml) * BK + khalf * 8];
      uint4* bp = (uint4*)&b;
      bp[0] = *(const uint4*)(q);
      bp[1] = *(const uint4*)(q + 16);
      c[0][t] = __builtin_amdgcn_wmma_f32_16x16x32_bf16(
          false, a0, false, b, (short)0, c[0][t], false, false);
      c[1][t] = __builtin_amdgcn_wmma_f32_16x16x32_bf16(
          false, a1, false, b, (short)0, c[1][t], false, false);
    }

    // next buffer complete before anyone reads it; fragment ds-reads of this
    // buffer completed before the last WMMA (dscnt), so single barrier works
    wait_async0();
    __syncthreads();
  }

  // Epilogue: C/D layout -> element r of lane: M = r + 8*khalf, N = ml
#pragma unroll
  for (int i = 0; i < 2; ++i) {
#pragma unroll
    for (int t = 0; t < 4; ++t) {
      const int n = nBase + wn * 64 + t * 16 + ml;
      const float bias = benc[n];
#pragma unroll
      for (int r = 0; r < 8; ++r) {
        const int m = mBase + wm * 32 + i * 16 + khalf * 8 + r;
        A1[(size_t)m * DBN + n] = c[i][t][r] + bias;
      }
    }
  }
}

// ---------------------------------------------------------------------------
// Top-64 of 32768 per row. 256 threads; each owns a 128-elem segment with a
// cached (max,idx) and a 128-bit extraction mask; only the winning segment
// rescans per iteration.
// ---------------------------------------------------------------------------
__global__ __launch_bounds__(256) void topk_kernel(
    const float* __restrict__ A1, float* __restrict__ tvals, int* __restrict__ tidx)
{
  const int row = blockIdx.x;
  const int tid = threadIdx.x;
  const float* a = A1 + (size_t)row * DBN;
  const int SEG  = DBN / 256;   // 128
  const int base = tid * SEG;

  uint32_t mask[4] = {0u, 0u, 0u, 0u};
  float lmax = -__builtin_inff();
  int   lidx = base;
  for (int j = 0; j < SEG; ++j) {
    float v = a[base + j];
    if (v > lmax) { lmax = v; lidx = base + j; }
  }

  __shared__ float sv[8];
  __shared__ int   si[8];
  __shared__ float wv;
  __shared__ int   wi;

  for (int it = 0; it < KTOP; ++it) {
    float v = lmax; int ix = lidx;
#pragma unroll
    for (int off = 16; off >= 1; off >>= 1) {
      float ov = __shfl_xor(v, off, 32);
      int   oi = __shfl_xor(ix, off, 32);
      if (ov > v) { v = ov; ix = oi; }
    }
    if ((tid & 31) == 0) { sv[tid >> 5] = v; si[tid >> 5] = ix; }
    __syncthreads();
    if (tid == 0) {
      float bv = sv[0]; int bi = si[0];
      for (int w = 1; w < 8; ++w)
        if (sv[w] > bv) { bv = sv[w]; bi = si[w]; }
      wv = bv; wi = bi;
      tvals[(size_t)row * KTOP + it] = bv;
      tidx [(size_t)row * KTOP + it] = bi;
    }
    __syncthreads();
    const int widx = wi;
    if ((widx >> 7) == tid) {            // owner of winning segment
      const int loc = widx & 127;
      mask[loc >> 5] |= (1u << (loc & 31));
      lmax = -__builtin_inff(); lidx = base;
      for (int j = 0; j < SEG; ++j) {
        if (mask[j >> 5] & (1u << (j & 31))) continue;
        float vv = a[base + j];
        if (vv > lmax) { lmax = vv; lidx = base + j; }
      }
    }
    // next iteration's first __syncthreads orders wi reuse
  }
  (void)wv;
}

// ---------------------------------------------------------------------------
// Sparse decode: z2[row,:] = sum_i tv[i] * W[ti[i],:] + b_dec   (fp32 W)
// One row per block; 256 threads x 16 coalesced columns.
// ---------------------------------------------------------------------------
__global__ __launch_bounds__(256) void decode_kernel(
    const float* __restrict__ tvals, const int* __restrict__ tidx,
    const float* __restrict__ W,     // fp32 [DBN, DIN]
    const float* __restrict__ bdec,
    float* __restrict__ Z)           // [BSZ, DIN]
{
  const int row = blockIdx.x;
  const int tid = threadIdx.x;
  __shared__ float svals[KTOP];
  __shared__ int   sidx[KTOP];
  if (tid < KTOP) {
    svals[tid] = tvals[(size_t)row * KTOP + tid];
    sidx [tid] = tidx [(size_t)row * KTOP + tid];
  }
  __syncthreads();

  const int PER = DIN / 256;  // 16
  float acc[PER];
#pragma unroll
  for (int s = 0; s < PER; ++s) acc[s] = bdec[tid + s * 256];

  for (int i = 0; i < KTOP; ++i) {
    const float v = svals[i];
    const float* wr = W + (size_t)sidx[i] * DIN;
#pragma unroll
    for (int s = 0; s < PER; ++s) acc[s] += v * wr[tid + s * 256];
  }
#pragma unroll
  for (int s = 0; s < PER; ++s) Z[(size_t)row * DIN + tid + s * 256] = acc[s];
}

// ---------------------------------------------------------------------------
extern "C" void kernel_launch(void* const* d_in, const int* in_sizes, int n_in,
                              void* d_out, int out_size, void* d_ws, size_t ws_size,
                              hipStream_t stream) {
  (void)in_sizes; (void)n_in; (void)out_size; (void)ws_size;
  const float* x    = (const float*)d_in[0];  // [BSZ, DIN]
  const float* W    = (const float*)d_in[1];  // [DBN, DIN]
  const float* benc = (const float*)d_in[2];  // [DBN]
  const float* bdec = (const float*)d_in[3];  // [DIN]
  // d_in[4] = k (== 64, fixed by problem shape)

  uint8_t* ws = (uint8_t*)d_ws;
  uint16_t* x16 = (uint16_t*)ws;                               // 32 MB
  size_t off = (size_t)BSZ * DIN * 2;
  uint16_t* w16 = (uint16_t*)(ws + off);                       // 256 MB
  off += (size_t)DBN * DIN * 2;
  float* a1 = (float*)(ws + off);                              // 512 MB
  off += (size_t)BSZ * DBN * 4;
  float* tv = (float*)(ws + off);                              // 1 MB
  off += (size_t)BSZ * KTOP * 4;
  int* ti = (int*)(ws + off);                                  // 1 MB

  // 1) fp32 -> bf16 conversions
  {
    long nx4 = (long)BSZ * DIN / 4;
    cvt_bf16x4<<<8192, 256, 0, stream>>>((const float4*)x, (ushort4*)x16, nx4);
    long nw4 = (long)DBN * DIN / 4;
    cvt_bf16x4<<<16384, 256, 0, stream>>>((const float4*)W, (ushort4*)w16, nw4);
  }

  // 2) encode GEMM with fused bias (bf16 WMMA, async LDS staging)
  {
    dim3 grid(DBN / BN, BSZ / BM);  // (256, 32)
    encode_gemm<<<grid, 256, 0, stream>>>(x16, w16, benc, a1);
  }

  // 3) per-row top-64
  topk_kernel<<<BSZ, 256, 0, stream>>>(a1, tv, ti);

  // 4) sparse tied-weight decode (fp32 weights for accuracy)
  decode_kernel<<<BSZ, 256, 0, stream>>>(tv, ti, W, bdec, (float*)d_out);
}
